// TriangleAttention_50294067036161
// MI455X (gfx1250) — compile-verified, matrix operand order checked by reading, not run on hardware
//
#include <hip/hip_runtime.h>
#include <hip/hip_bf16.h>
#include <math.h>

// ---------------------------------------------------------------------------
// TriangleAttention (starting node) for gfx1250 / MI455X.
// bf16 WMMA (v_wmma_f32_16x16x32_bf16) for all GEMMs, f32 softmax.
// Weights pre-packed to bf16 in B-fragment lane order (2x b128 per fragment).
// ---------------------------------------------------------------------------

typedef __attribute__((ext_vector_type(16))) __bf16 v16bf;
typedef __attribute__((ext_vector_type(8)))  __bf16 v8bf;
typedef __attribute__((ext_vector_type(8)))  float  v8f;

#define NRES 256
#define CZ   128
#define NH   4
#define CH   32
#define HC   (NH*CH)   // 128

#define NPROJ_FRAG (33*4)   // 33 N-tiles x 4 K-slices
#define NOUT_FRAG  (8*4)    // 8 N-tiles x 4 K-slices

// Load a 16-bit A/B WMMA fragment from a contiguous 32-element K-row.
// Per ISA layout: lane half==0 holds K {0..7,16..23}, half==1 holds {8..15,24..31}.
__device__ __forceinline__ v16bf load_frag_b16(const __bf16* rowbase, int half) {
  const v8bf a = *(const v8bf*)(rowbase + half * 8);
  const v8bf b = *(const v8bf*)(rowbase + 16 + half * 8);
  v16bf r;
#pragma unroll
  for (int t = 0; t < 8; ++t) { r[t] = a[t]; r[8 + t] = b[t]; }
  return r;
}

__device__ __forceinline__ v8f wmma_bf16(v16bf a, v16bf b, v8f c) {
  return __builtin_amdgcn_wmma_f32_16x16x32_bf16(false, a, false, b, (short)0, c, false, false);
}

// ---------------------------------------------------------------------------
// Kernel 0: pre-pack fp32 weights into bf16 B-fragment lane order.
// wpack:  [frag][lane][16]  frag = tile*4+kk, tile over [q|k|v|g|bias] cols
// wopack: [frag][lane][16]  for Wo
// Attention scale 1/sqrt(C) folded into Wq.
// ---------------------------------------------------------------------------
__global__ __launch_bounds__(256) void k_prepack(
    const float* __restrict__ Wb, const float* __restrict__ Wq, const float* __restrict__ Wk,
    const float* __restrict__ Wv, const float* __restrict__ Wg, const float* __restrict__ Wo,
    __bf16* __restrict__ wpack, __bf16* __restrict__ wopack)
{
  const int id = blockIdx.x * blockDim.x + threadIdx.x;
  const float qscale = 0.17677669529663689f;  // 1/sqrt(32)
  if (id < NPROJ_FRAG * 32) {
    const int lane = id & 31, frag = id >> 5;
    const int kk = frag & 3, tile = frag >> 2;
    const int n = lane & 15, half = lane >> 4;
    const int col = tile * 16 + n;
    __bf16* dst = wpack + (size_t)frag * 512 + lane * 16;
#pragma unroll
    for (int t = 0; t < 16; ++t) {
      const int krow = kk * 32 + ((t < 8) ? (half * 8 + t) : (16 + half * 8 + (t - 8)));
      float w;
      if (col < 128)      w = Wq[krow * HC + col] * qscale;
      else if (col < 256) w = Wk[krow * HC + col - 128];
      else if (col < 384) w = Wv[krow * HC + col - 256];
      else if (col < 512) w = Wg[krow * HC + col - 384];
      else if (col < 516) w = Wb[krow * NH + col - 512];
      else                w = 0.f;
      dst[t] = (__bf16)w;
    }
  } else if (id < (NPROJ_FRAG + NOUT_FRAG) * 32) {
    const int id2 = id - NPROJ_FRAG * 32;
    const int lane = id2 & 31, frag = id2 >> 5;
    const int kk = frag & 3, tile = frag >> 2;
    const int n = lane & 15, half = lane >> 4;
    const int col = tile * 16 + n;
    __bf16* dst = wopack + (size_t)frag * 512 + lane * 16;
#pragma unroll
    for (int t = 0; t < 16; ++t) {
      const int krow = kk * 32 + ((t < 8) ? (half * 8 + t) : (16 + half * 8 + (t - 8)));
      dst[t] = (__bf16)Wo[krow * CZ + col];
    }
  }
}

// ---------------------------------------------------------------------------
// Kernel 1: LayerNorm + fused q/k/v/g/bias projection (16-row GEMM tile/block)
// ---------------------------------------------------------------------------
__global__ __launch_bounds__(128) void k_ln_proj(
    const float* __restrict__ z, const float* __restrict__ lng, const float* __restrict__ lnb,
    const float* __restrict__ bg, const __bf16* __restrict__ wpack,
    __bf16* __restrict__ qws, __bf16* __restrict__ kws, __bf16* __restrict__ vtws,
    __bf16* __restrict__ gws, float* __restrict__ bws)
{
  __shared__ __bf16 xs[16 * CZ];
  const int tid  = threadIdx.x;
  const int lane = tid & 31, half = lane >> 4, n = lane & 15;
  const int wave = tid >> 5;
  const int row0 = blockIdx.x * 16;

  // ---- LayerNorm: 16 rows x 128, 8 threads per row ----
  {
    const int r   = tid >> 3;   // 0..15
    const int seg = tid & 7;    // 0..7
    const float* src = z + (size_t)(row0 + r) * CZ + seg * 16;
    float v[16];
#pragma unroll
    for (int t = 0; t < 4; ++t) {
      float4 f = ((const float4*)src)[t];
      v[t*4+0] = f.x; v[t*4+1] = f.y; v[t*4+2] = f.z; v[t*4+3] = f.w;
    }
    float s = 0.f, sq = 0.f;
#pragma unroll
    for (int t = 0; t < 16; ++t) { s += v[t]; sq += v[t] * v[t]; }
#pragma unroll
    for (int m = 4; m >= 1; m >>= 1) {
      s  += __shfl_xor(s,  m, 8);
      sq += __shfl_xor(sq, m, 8);
    }
    const float mean = s * (1.f / CZ);
    const float var  = sq * (1.f / CZ) - mean * mean;
    const float rstd = rsqrtf(var + 1e-5f);
#pragma unroll
    for (int t = 0; t < 16; ++t) {
      const int c = seg * 16 + t;
      xs[r * CZ + c] = (__bf16)((v[t] - mean) * rstd * lng[c] + lnb[c]);
    }
  }
  __syncthreads();

  // A fragments: 16 rows x 128 K, as 4 K-slices of 32
  v16bf afr[4];
#pragma unroll
  for (int kk = 0; kk < 4; ++kk) afr[kk] = load_frag_b16(&xs[n * CZ + kk * 32], half);

  // 33 output N-tiles of 16: [0,128)q [128,256)k [256,384)v [384,512)g [512,516)bias
  for (int tile = wave; tile < 33; tile += 4) {
    const int col = tile * 16 + n;
    v8f acc = {};
#pragma unroll
    for (int kk = 0; kk < 4; ++kk) {
      const v16bf bfr = *(const v16bf*)(wpack + ((size_t)(tile * 4 + kk)) * 512 + (size_t)lane * 16);
      acc = wmma_bf16(afr[kk], bfr, acc);
    }
    // Scatter D fragment: value (m,n) lives at lane = 16*(m/8)+n, vgpr r=m%8
#pragma unroll
    for (int r = 0; r < 8; ++r) {
      const int m = r + 8 * half;
      const int R = row0 + m;
      const int i = R >> 8, j = R & (NRES - 1);
      const float val = acc[r];
      if (tile < 8) {
        const int h = col >> 5, c = col & 31;
        qws[(((size_t)i * NH + h) * NRES + j) * CH + c] = (__bf16)val;   // q pre-scaled
      } else if (tile < 16) {
        const int c2 = col - 128, h = c2 >> 5, c = c2 & 31;
        kws[(((size_t)i * NH + h) * NRES + j) * CH + c] = (__bf16)val;
      } else if (tile < 24) {
        const int c2 = col - 256, h = c2 >> 5, c = c2 & 31;
        vtws[(((size_t)i * NH + h) * CH + c) * NRES + j] = (__bf16)val;  // transposed
      } else if (tile < 32) {
        const int c2 = col - 384;
        // sigmoid via v_rcp_f32 (no IEEE divide sequence)
        const float gv = __builtin_amdgcn_rcpf(1.f + __expf(-(val + bg[c2])));
        gws[(size_t)R * HC + c2] = (__bf16)gv;
      } else if (col < 516) {
        bws[(size_t)(col - 512) * NRES * NRES + R] = val;  // b[h][j][k]
      }
    }
  }
}

// ---------------------------------------------------------------------------
// Kernel 2: flash attention per (i, h); 8 waves x 32 queries; key tiles of 32
// ---------------------------------------------------------------------------
__global__ __launch_bounds__(256) void k_attn(
    const __bf16* __restrict__ qws, const __bf16* __restrict__ kws,
    const __bf16* __restrict__ vtws, const __bf16* __restrict__ gws,
    const float* __restrict__ bws, __bf16* __restrict__ ows)
{
  __shared__ __bf16 pls[8][16 * 32];   // wave-private P staging (re-layout)
  const int i = blockIdx.x, h = blockIdx.y;
  const int tid  = threadIdx.x;
  const int lane = tid & 31, half = lane >> 4, n = lane & 15;
  const int wave = tid >> 5;
  const int qbase = wave * 32;

  const __bf16* qp  = qws  + ((size_t)i * NH + h) * NRES * CH;
  const __bf16* kp  = kws  + ((size_t)i * NH + h) * NRES * CH;
  const __bf16* vtp = vtws + ((size_t)i * NH + h) * CH * NRES;
  const float*  bp  = bws  + (size_t)h * NRES * NRES;

  v16bf aq[2];
#pragma unroll
  for (int t = 0; t < 2; ++t)
    aq[t] = load_frag_b16(qp + (size_t)(qbase + t * 16 + n) * CH, half);

  float mx[2][8], sm[2][8];
  v8f oacc[2][2];
  v8f vzero = {};
#pragma unroll
  for (int t = 0; t < 2; ++t) {
#pragma unroll
    for (int r = 0; r < 8; ++r) { mx[t][r] = -1e30f; sm[t][r] = 0.f; }
    oacc[t][0] = vzero; oacc[t][1] = vzero;
  }

  for (int kt = 0; kt < 8; ++kt) {
    const int kb = kt * 32;
    if (kt < 7) __builtin_prefetch(kp + (size_t)(kb + 32) * CH, 0, 0);  // global_prefetch_b8

    v16bf bk[2], bv[2];
#pragma unroll
    for (int nn = 0; nn < 2; ++nn)
      bk[nn] = load_frag_b16(kp + (size_t)(kb + nn * 16 + n) * CH, half);
#pragma unroll
    for (int cc = 0; cc < 2; ++cc)
      bv[cc] = load_frag_b16(vtp + (size_t)(cc * 16 + n) * NRES + kb, half);

#pragma unroll
    for (int t = 0; t < 2; ++t) {
      v8f s0 = wmma_bf16(aq[t], bk[0], vzero);
      v8f s1 = wmma_bf16(aq[t], bk[1], vzero);
      float p0[8], p1[8], alpha[8];
#pragma unroll
      for (int r = 0; r < 8; ++r) {
        const int m = qbase + t * 16 + r + 8 * half;
        float x0 = s0[r] + bp[(size_t)m * NRES + kb + n];        // scale folded into q
        float x1 = s1[r] + bp[(size_t)m * NRES + kb + 16 + n];
        float tm = fmaxf(x0, x1);
#pragma unroll
        for (int off = 8; off >= 1; off >>= 1) tm = fmaxf(tm, __shfl_xor(tm, off, 16));
        const float nm = fmaxf(mx[t][r], tm);
        alpha[r] = __expf(mx[t][r] - nm);
        mx[t][r] = nm;
        p0[r] = __expf(x0 - nm);
        p1[r] = __expf(x1 - nm);
        float ps = p0[r] + p1[r];
#pragma unroll
        for (int off = 8; off >= 1; off >>= 1) ps += __shfl_xor(ps, off, 16);
        sm[t][r] = sm[t][r] * alpha[r] + ps;
      }
#pragma unroll
      for (int cc = 0; cc < 2; ++cc)
#pragma unroll
        for (int r = 0; r < 8; ++r) oacc[t][cc][r] *= alpha[r];

      // P (D-layout f32) -> LDS bf16 -> A-layout fragment
      __bf16* pw = &pls[wave][0];
#pragma unroll
      for (int r = 0; r < 8; ++r) {
        const int ml = r + 8 * half;
        pw[ml * 32 + n]      = (__bf16)p0[r];
        pw[ml * 32 + 16 + n] = (__bf16)p1[r];
      }
      asm volatile("s_wait_dscnt 0" ::: "memory");
      const v16bf ap = load_frag_b16(pw + n * 32, half);
#pragma unroll
      for (int cc = 0; cc < 2; ++cc) oacc[t][cc] = wmma_bf16(ap, bv[cc], oacc[t][cc]);
    }
  }

  // Epilogue: normalize (v_rcp_f32), gate, store bf16 (row-major (i*N+q, h*C+c))
#pragma unroll
  for (int t = 0; t < 2; ++t)
#pragma unroll
    for (int r = 0; r < 8; ++r) {
      const float inv = __builtin_amdgcn_rcpf(sm[t][r]);
      const int m = qbase + t * 16 + r + 8 * half;
      const size_t rowg = (size_t)i * NRES + m;
#pragma unroll
      for (int cc = 0; cc < 2; ++cc) {
        const int colh = h * CH + cc * 16 + n;
        float val = oacc[t][cc][r] * inv;
        val *= (float)gws[rowg * HC + colh];
        ows[rowg * HC + colh] = (__bf16)val;
      }
    }
}

// ---------------------------------------------------------------------------
// Kernel 3: output projection (65536 x 128) x (128 x 128) + bo -> f32
// ---------------------------------------------------------------------------
__global__ __launch_bounds__(128) void k_out(
    const __bf16* __restrict__ ows, const __bf16* __restrict__ wopack,
    const float* __restrict__ bo, float* __restrict__ out)
{
  const int tid  = threadIdx.x;
  const int lane = tid & 31, half = lane >> 4, n = lane & 15;
  const int wave = tid >> 5;
  const int row0 = blockIdx.x * 16;

  v16bf afr[4];
#pragma unroll
  for (int kk = 0; kk < 4; ++kk)
    afr[kk] = load_frag_b16(ows + (size_t)(row0 + n) * HC + kk * 32, half);

  for (int tile = wave; tile < 8; tile += 4) {
    const int col = tile * 16 + n;
    v8f acc = {};
#pragma unroll
    for (int kk = 0; kk < 4; ++kk) {
      const v16bf bfr = *(const v16bf*)(wopack + ((size_t)(tile * 4 + kk)) * 512 + (size_t)lane * 16);
      acc = wmma_bf16(afr[kk], bfr, acc);
    }
    const float bias = bo[col];
#pragma unroll
    for (int r = 0; r < 8; ++r) {
      const int m = r + 8 * half;
      out[(size_t)(row0 + m) * CZ + col] = acc[r] + bias;
    }
  }
}

// ---------------------------------------------------------------------------
extern "C" void kernel_launch(void* const* d_in, const int* in_sizes, int n_in,
                              void* d_out, int out_size, void* d_ws, size_t ws_size,
                              hipStream_t stream) {
  const float* z   = (const float*)d_in[0];
  const float* lng = (const float*)d_in[1];
  const float* lnb = (const float*)d_in[2];
  const float* Wb  = (const float*)d_in[3];
  const float* Wq  = (const float*)d_in[4];
  const float* Wk  = (const float*)d_in[5];
  const float* Wv  = (const float*)d_in[6];
  const float* Wg  = (const float*)d_in[7];
  const float* bg  = (const float*)d_in[8];
  const float* Wo  = (const float*)d_in[9];
  const float* bo  = (const float*)d_in[10];
  float* out = (float*)d_out;

  char* ws = (char*)d_ws;
  const size_t SZ = (size_t)NRES * NRES * HC;        // 8M elements
  __bf16* qws  = (__bf16*)(ws);                      // 16 MB  (i,h,j,c), scale folded
  __bf16* kws  = (__bf16*)(ws + SZ * 2);             // 16 MB  (i,h,j,c)
  __bf16* vtws = (__bf16*)(ws + SZ * 4);             // 16 MB  (i,h,c,j)
  __bf16* gws  = (__bf16*)(ws + SZ * 6);             // 16 MB  (iq, hc) sigmoid applied
  __bf16* ows  = (__bf16*)(ws + SZ * 8);             // 16 MB  (iq, hc)
  float*  bws  = (float*)(ws + SZ * 10);             // 1 MB   (h, j, k)
  __bf16* wpack  = (__bf16*)(ws + SZ * 10 + (1u << 20));                  // 132 KB
  __bf16* wopack = (__bf16*)(ws + SZ * 10 + (1u << 20) + NPROJ_FRAG*512*2); // 32 KB

  k_prepack<<<dim3(21), dim3(256), 0, stream>>>(Wb, Wq, Wk, Wv, Wg, Wo, wpack, wopack);
  k_ln_proj<<<dim3(4096), dim3(128), 0, stream>>>(z, lng, lnb, bg, wpack,
                                                  qws, kws, vtws, gws, bws);
  k_attn<<<dim3(256, 4), dim3(256), 0, stream>>>(qws, kws, vtws, gws, bws, ows);
  k_out<<<dim3(4096), dim3(128), 0, stream>>>(ows, wopack, bo, out);
}
